// CausalSelfAttention_88364657148333
// MI455X (gfx1250) — matrix operand order repaired and checked by reading, not compiled, not run on hardware
//
#include <hip/hip_runtime.h>
#include <stdint.h>

#define S_LEN 2048
#define DIM   2048
#define NH    16
#define NKV   4
#define DH    128
#define BATCH 2
#define GROUP (NH / NKV)

typedef __attribute__((ext_vector_type(16))) __bf16 bf16x16;
typedef __attribute__((ext_vector_type(8)))  float  f32x8;
typedef __attribute__((ext_vector_type(4)))  uint32_t u32x4;
typedef __attribute__((ext_vector_type(8)))  int      i32x8;
typedef __attribute__((ext_vector_type(4)))  int      i32x4;

union FragAB { uint4 u[2]; bf16x16 v; };

__device__ __forceinline__ uint16_t f32_to_bf16(float f) {
  union { float f; uint32_t u; } c; c.f = f;
  uint32_t u = c.u;
  u += 0x7FFFu + ((u >> 16) & 1u);   // round-to-nearest-even
  return (uint16_t)(u >> 16);
}

__device__ __forceinline__ f32x8 wmma_bf16(FragAB a, FragAB b, f32x8 c) {
  return __builtin_amdgcn_wmma_f32_16x16x32_bf16(false, a.v, false, b.v,
                                                 (short)0, c, false, false);
}

// TDM: DMA one 64-row x 32-col bf16 tile (row stride = rowstride elems) into LDS.
// D# per ISA 08 §8; 6-arg builtin form (amdgpu-toolchain clang-23).
__device__ __forceinline__ void tdm_load_tile_64x32(uint32_t lds_off,
                                                    const uint16_t* gsrc,
                                                    int rowstride, int nrows) {
  uint64_t g = (uint64_t)(uintptr_t)gsrc;
  // group0: count=1 | lds_addr | global_addr[56:0] | type=2
  u32x4 g0 = { 1u, lds_off, (uint32_t)g,
               (uint32_t)((g >> 32) & 0x01FFFFFFull) | (2u << 30) };
  // group1: data_size=2B; tensor_dim0=rowstride, tensor_dim1=nrows;
  //         tile_dim0=32, tile_dim1=64; tensor_dim0_stride=rowstride
  i32x8 g1 = { (int)0x00010000u,
               (int)(((unsigned)rowstride & 0xFFFFu) << 16),
               (int)(((unsigned)rowstride >> 16) | (((unsigned)nrows & 0xFFFFu) << 16)),
               (int)(((unsigned)nrows >> 16) | (32u << 16)),
               (int)64,
               (int)rowstride,
               0, 0 };
  i32x4 z4 = { 0, 0, 0, 0 };
  i32x8 z8 = { 0, 0, 0, 0, 0, 0, 0, 0 };
  __builtin_amdgcn_tensor_load_to_lds(g0, g1, z4, z4, z8, 0);
}

// ---------------------------------------------------------------- fp32 -> bf16
__global__ __launch_bounds__(256) void cvt_kernel(const float* __restrict__ in,
                                                  uint16_t* __restrict__ out,
                                                  size_t n) {
  size_t i = (size_t)blockIdx.x * 256 + threadIdx.x;
  if (i < n) out[i] = f32_to_bf16(in[i]);
}

// ---------------------------------------------------------------- GEMM
// C[m,n] = sum_k A[m,k] * W[n,k]  (A: MxK bf16, W: NxK bf16, both row-major)
// Block = 8 waves: wave w -> 16 M-rows (block tile 128M x 64N).
// The shared 64x32 B tile is DMA'd into LDS by the TDM, double-buffered:
// wave 0 issues the next tile's tensor_load_to_lds before the block computes
// on the current tile, waits TENSORcnt after compute, then block-barrier.
__global__ __launch_bounds__(256) void gemm_bf16(const uint16_t* __restrict__ A,
                                                 const uint16_t* __restrict__ W,
                                                 float* __restrict__ C,
                                                 int M, int N, int K) {
  (void)M;
  const int lane = threadIdx.x & 31;
  const int wave = threadIdx.x >> 5;
  const int lh   = lane >> 4;     // lane-half
  const int ll   = lane & 15;
  const int m0   = (blockIdx.y * 8 + wave) * 16;
  const int n0   = blockIdx.x * 64;

  __shared__ __align__(16) uint16_t bsm[2][64 * 32];   // 2 x 4KB

  f32x8 acc[4] = {};
  const uint16_t* arow = A + (size_t)(m0 + ll) * K;
  const uint16_t* wtile = W + (size_t)n0 * K;
  const uint32_t lds0 = (uint32_t)(uintptr_t)&bsm[0][0];
  const uint32_t lds1 = (uint32_t)(uintptr_t)&bsm[1][0];

  const int NK = K >> 5;
  if (wave == 0) {
    tdm_load_tile_64x32(lds0, wtile, K, N);
    __builtin_amdgcn_s_wait_tensorcnt(0);
  }
  __syncthreads();

  for (int kc = 0; kc < NK; ++kc) {
    if (wave == 0 && kc + 1 < NK)
      tdm_load_tile_64x32(((kc + 1) & 1) ? lds1 : lds0,
                          wtile + (size_t)(kc + 1) * 32, K, N);

    FragAB a;
    const uint16_t* ak = arow + kc * 32;
    a.u[0] = *(const uint4*)(ak + lh * 8);
    a.u[1] = *(const uint4*)(ak + 16 + lh * 8);

    const uint16_t* bb = &bsm[kc & 1][0];
    FragAB b[4];
#pragma unroll
    for (int t = 0; t < 4; ++t) {   // B 32x16: lane = column (tile row n)
      const uint16_t* brow = bb + (t * 16 + ll) * 32 + lh * 8;
      b[t].u[0] = *(const uint4*)(brow);
      b[t].u[1] = *(const uint4*)(brow + 16);
    }
#pragma unroll
    for (int t = 0; t < 4; ++t) acc[t] = wmma_bf16(a, b[t], acc[t]);

    if (wave == 0 && kc + 1 < NK) __builtin_amdgcn_s_wait_tensorcnt(0);
    __syncthreads();
  }
#pragma unroll
  for (int t = 0; t < 4; ++t)
#pragma unroll
    for (int r = 0; r < 8; ++r)
      C[(size_t)(m0 + r + 8 * lh) * N + n0 + t * 16 + ll] = acc[t][r];
}

// ---------------------------------------------------------------- gate
__global__ __launch_bounds__(256) void gate_kernel(const float* __restrict__ x,
                                                   const float* __restrict__ gw,
                                                   const float* __restrict__ gb,
                                                   float* __restrict__ gate) {
  __shared__ float red[256 * NH];
  const int row = blockIdx.x;           // b*S + s
  const int t = threadIdx.x;
  float p[NH];
#pragma unroll
  for (int h = 0; h < NH; ++h) p[h] = 0.f;
  const float* xr = x + (size_t)row * DIM;
  for (int e = t; e < DIM; e += 256) {
    float xv = xr[e];
#pragma unroll
    for (int h = 0; h < NH; ++h) p[h] += xv * gw[h * DIM + e];
  }
#pragma unroll
  for (int h = 0; h < NH; ++h) red[t * NH + h] = p[h];
  __syncthreads();
  if (t < NH) {
    float s = 0.f;
    for (int i = 0; i < 256; ++i) s += red[i * NH + t];
    s += gb[t];
    gate[(size_t)row * NH + t] = 1.f / (1.f + __expf(-s));
  }
}

// ---------------------------------------------------------------- RMSNorm + RoPE
// in:  [B,S,Hn,D] fp32    out: [B,Hn,S,D] bf16
__global__ __launch_bounds__(128) void norm_rope_kernel(const float* __restrict__ in,
                                                        uint16_t* __restrict__ outbf,
                                                        const float* __restrict__ gain,
                                                        int Hn, int use_gain) {
  const int row = blockIdx.x;
  const int d = threadIdx.x;
  const int h = row % Hn;
  const int s = (row / Hn) % S_LEN;
  const int b = row / (Hn * S_LEN);
  __shared__ float xs[DH];
  __shared__ float red[DH];

  float xv = in[(size_t)row * DH + d];
  red[d] = xv * xv;
  __syncthreads();
  for (int off = 64; off > 0; off >>= 1) {
    if (d < off) red[d] += red[d + off];
    __syncthreads();
  }
  float r = rsqrtf(red[0] * (1.0f / DH) + 1.1920929e-07f);
  float xn = xv * r;
  xs[d] = xn;
  __syncthreads();

  float out;
  if (d < 32) {
    float inv_freq = __powf(10000.0f, -(float)(2 * d) / 64.0f);
    float fr = (float)s * inv_freq;
    out = xn * __cosf(fr) + xs[d + 32] * __sinf(fr);
  } else if (d < 64) {
    int i = d - 32;
    float inv_freq = __powf(10000.0f, -(float)(2 * i) / 64.0f);
    float fr = (float)s * inv_freq;
    out = -xs[i] * __sinf(fr) + xn * __cosf(fr);
  } else {
    out = xn;
  }
  if (use_gain) out *= gain[h];
  outbf[(((size_t)b * Hn + h) * S_LEN + s) * DH + d] = f32_to_bf16(out);
}

// ---------------------------------------------------------------- V mix
// Vtmp: [B,S,HKV,D] fp32; v0: [B,HKV,S,D]; rawv: [B,HKV,S,D] fp32
// Vt:   [B,HKV,D,S] bf16 (pre-transposed so P@V B-fragments are contiguous)
__global__ __launch_bounds__(256) void vmix_kernel(const float* __restrict__ Vtmp,
                                                   const float* __restrict__ v0,
                                                   const float* __restrict__ lam,
                                                   float* __restrict__ rawv,
                                                   uint16_t* __restrict__ Vt) {
  size_t i = (size_t)blockIdx.x * 256 + threadIdx.x;  // over B*HKV*S*D = 2^21
  int d = (int)(i & (DH - 1));
  int s = (int)((i >> 7) & (S_LEN - 1));
  int hkv = (int)((i >> 18) & (NKV - 1));
  int b = (int)(i >> 20);
  float vt = Vtmp[(((size_t)b * S_LEN + s) * NKV + hkv) * DH + d];
  rawv[i] = vt;
  float vm = lam[0] * v0[i] + lam[1] * vt;
  Vt[(((size_t)b * NKV + hkv) * DH + d) * S_LEN + s] = f32_to_bf16(vm);
}

// ---------------------------------------------------------------- flash attention
// Qbf: [B,NH,S,D]  Kbf: [B,NKV,S,D]  Vt: [B,NKV,D,S]  gate: [B,S,NH]
// Ybf: [B,S,NH*D] bf16.  One wave = one 16-row q tile.
__global__ __launch_bounds__(128) void attn_kernel(const uint16_t* __restrict__ Qbf,
                                                   const uint16_t* __restrict__ Kbf,
                                                   const uint16_t* __restrict__ Vt,
                                                   const float* __restrict__ gate,
                                                   uint16_t* __restrict__ Ybf) {
  const int lane = threadIdx.x & 31;
  const int wave = threadIdx.x >> 5;
  const int lh = lane >> 4, ll = lane & 15;
  const int qtile = blockIdx.x * 4 + wave;
  const int h = blockIdx.y;
  const int b = blockIdx.z;
  const int hkv = h / GROUP;
  const int qbase = qtile * 16;
  const float scale = 0.08838834764831845f;  // 1/sqrt(128)

  __shared__ __align__(16) uint16_t plds_all[4][16 * 32];
  uint16_t* plds = plds_all[wave];

  FragAB qf[4];
  const uint16_t* qrow = Qbf + (((size_t)b * NH + h) * S_LEN + qbase + ll) * DH;
#pragma unroll
  for (int j = 0; j < 4; ++j) {
    qf[j].u[0] = *(const uint4*)(qrow + j * 32 + lh * 8);
    qf[j].u[1] = *(const uint4*)(qrow + j * 32 + 16 + lh * 8);
  }

  f32x8 acc[8] = {};
  float mrow[8], lrow[8], alpha[8];
#pragma unroll
  for (int r = 0; r < 8; ++r) { mrow[r] = -3.0e38f; lrow[r] = 0.f; }

  const uint16_t* kbase = Kbf + ((size_t)b * NKV + hkv) * S_LEN * DH;
  const uint16_t* vtb = Vt + ((size_t)b * NKV + hkv) * (size_t)DH * S_LEN;
  const int nkt = qtile / 2 + 1;

  for (int kt = 0; kt < nkt; ++kt) {
    const int k0 = kt * 32;
    f32x8 s0 = {}, s1 = {};
#pragma unroll
    for (int j = 0; j < 4; ++j) {
      FragAB kb0, kb1;
      const uint16_t* kr0 = kbase + (size_t)(k0 + ll) * DH + j * 32 + lh * 8;
      const uint16_t* kr1 = kbase + (size_t)(k0 + 16 + ll) * DH + j * 32 + lh * 8;
      kb0.u[0] = *(const uint4*)(kr0);
      kb0.u[1] = *(const uint4*)(kr0 + 16);
      kb1.u[0] = *(const uint4*)(kr1);
      kb1.u[1] = *(const uint4*)(kr1 + 16);
      s0 = wmma_bf16(qf[j], kb0, s0);
      s1 = wmma_bf16(qf[j], kb1, s1);
    }
#pragma unroll
    for (int r = 0; r < 8; ++r) {
      const int qi = qbase + r + 8 * lh;
      float v0s = s0[r] * scale;
      float v1s = s1[r] * scale;
      if (k0 + ll > qi) v0s = -3.0e38f;
      if (k0 + 16 + ll > qi) v1s = -3.0e38f;
      float mx = fmaxf(v0s, v1s);
#pragma unroll
      for (int off = 1; off < 16; off <<= 1) mx = fmaxf(mx, __shfl_xor(mx, off, 32));
      float mnew = fmaxf(mrow[r], mx);
      float al = __expf(mrow[r] - mnew);
      float p0 = __expf(v0s - mnew);
      float p1 = __expf(v1s - mnew);
      float rs = p0 + p1;
#pragma unroll
      for (int off = 1; off < 16; off <<= 1) rs += __shfl_xor(rs, off, 32);
      lrow[r] = lrow[r] * al + rs;
      mrow[r] = mnew;
      alpha[r] = al;
      plds[(r + 8 * lh) * 32 + ll] = f32_to_bf16(p0);
      plds[(r + 8 * lh) * 32 + 16 + ll] = f32_to_bf16(p1);
    }
#pragma unroll
    for (int t = 0; t < 8; ++t)
#pragma unroll
      for (int r = 0; r < 8; ++r) acc[t][r] *= alpha[r];

    asm volatile("s_wait_dscnt 0" ::: "memory");

    FragAB pf;
    pf.u[0] = *(const uint4*)(plds + ll * 32 + lh * 8);
    pf.u[1] = *(const uint4*)(plds + ll * 32 + 16 + lh * 8);

#pragma unroll
    for (int j = 0; j < 8; ++j) {
      FragAB vf;
      const uint16_t* vr = vtb + (size_t)(j * 16 + ll) * S_LEN + k0 + lh * 8;
      vf.u[0] = *(const uint4*)(vr);
      vf.u[1] = *(const uint4*)(vr + 16);
      acc[j] = wmma_bf16(pf, vf, acc[j]);
    }
  }

#pragma unroll
  for (int r = 0; r < 8; ++r) {
    const int qi = qbase + r + 8 * lh;
    const float g = gate[((size_t)b * S_LEN + qi) * NH + h];
    const float sc = g / lrow[r];
    const size_t orow = ((size_t)b * S_LEN + qi) * (size_t)DIM + h * DH;
#pragma unroll
    for (int t = 0; t < 8; ++t)
      Ybf[orow + t * 16 + ll] = f32_to_bf16(acc[t][r] * sc);
  }
}

// ---------------------------------------------------------------- launch
extern "C" void kernel_launch(void* const* d_in, const int* in_sizes, int n_in,
                              void* d_out, int out_size, void* d_ws, size_t ws_size,
                              hipStream_t stream) {
  (void)in_sizes; (void)n_in; (void)out_size; (void)ws_size;
  const float* x   = (const float*)d_in[0];
  const float* v0  = (const float*)d_in[1];
  const float* Wq  = (const float*)d_in[2];
  const float* Wk  = (const float*)d_in[3];
  const float* Wv  = (const float*)d_in[4];
  const float* Wp  = (const float*)d_in[5];
  const float* qg  = (const float*)d_in[6];
  const float* lam = (const float*)d_in[7];
  const float* gw  = (const float*)d_in[8];
  const float* gb  = (const float*)d_in[9];

  float* out  = (float*)d_out;                               // [B,S,DIM]
  float* rawv = out + (size_t)BATCH * S_LEN * DIM;           // [B,NKV,S,DH]

  char* ws = (char*)d_ws;
  size_t off = 0;
  auto take = [&](size_t bytes) -> void* {
    void* p = ws + off;
    off += (bytes + 255) & ~(size_t)255;
    return p;
  };
  const size_t MROWS = (size_t)BATCH * S_LEN;                // 4096
  uint16_t* Xbf  = (uint16_t*)take(MROWS * DIM * 2);
  uint16_t* Wqbf = (uint16_t*)take((size_t)DIM * DIM * 2);
  uint16_t* Wkbf = (uint16_t*)take((size_t)NKV * DH * DIM * 2);
  uint16_t* Wvbf = (uint16_t*)take((size_t)NKV * DH * DIM * 2);
  uint16_t* Wpbf = (uint16_t*)take((size_t)DIM * DIM * 2);
  float*    Qtmp = (float*)take(MROWS * DIM * 4);
  float*    Ktmp = (float*)take(MROWS * NKV * DH * 4);
  float*    Vtmp = (float*)take(MROWS * NKV * DH * 4);
  uint16_t* Qbf  = (uint16_t*)take(MROWS * DIM * 2);
  uint16_t* Kbf  = (uint16_t*)take(MROWS * NKV * DH * 2);
  uint16_t* Vtbf = (uint16_t*)take(MROWS * NKV * DH * 2);
  float*    gatep = (float*)take(MROWS * NH * 4);
  uint16_t* Ybf = (uint16_t*)Qtmp;  // alias: Qtmp fp32 dead after norm_rope

  auto cvt = [&](const float* src, uint16_t* dst, size_t n) {
    cvt_kernel<<<(unsigned)((n + 255) / 256), 256, 0, stream>>>(src, dst, n);
  };
  cvt(x, Xbf, MROWS * DIM);
  cvt(Wq, Wqbf, (size_t)DIM * DIM);
  cvt(Wk, Wkbf, (size_t)NKV * DH * DIM);
  cvt(Wv, Wvbf, (size_t)NKV * DH * DIM);
  cvt(Wp, Wpbf, (size_t)DIM * DIM);

  // QKV projections (TDM-staged B tiles)
  gemm_bf16<<<dim3(DIM / 64, (unsigned)(MROWS / 128)), 256, 0, stream>>>(
      Xbf, Wqbf, Qtmp, (int)MROWS, DIM, DIM);
  gemm_bf16<<<dim3((NKV * DH) / 64, (unsigned)(MROWS / 128)), 256, 0, stream>>>(
      Xbf, Wkbf, Ktmp, (int)MROWS, NKV * DH, DIM);
  gemm_bf16<<<dim3((NKV * DH) / 64, (unsigned)(MROWS / 128)), 256, 0, stream>>>(
      Xbf, Wvbf, Vtmp, (int)MROWS, NKV * DH, DIM);

  gate_kernel<<<(unsigned)MROWS, 256, 0, stream>>>(x, gw, gb, gatep);

  norm_rope_kernel<<<(unsigned)(MROWS * NH), 128, 0, stream>>>(Qtmp, Qbf, qg, NH, 1);
  norm_rope_kernel<<<(unsigned)(MROWS * NKV), 128, 0, stream>>>(Ktmp, Kbf, qg, NKV, 0);

  vmix_kernel<<<(unsigned)((MROWS * NKV * DH) / 256), 256, 0, stream>>>(
      Vtmp, v0, lam, rawv, Vtbf);

  attn_kernel<<<dim3(S_LEN / 64, NH, BATCH), 128, 0, stream>>>(
      Qbf, Kbf, Vtbf, gatep, Ybf);

  gemm_bf16<<<dim3(DIM / 64, (unsigned)(MROWS / 128)), 256, 0, stream>>>(
      Ybf, Wpbf, out, (int)MROWS, DIM, DIM);
}
